// kMeans_57174604645136
// MI455X (gfx1250) — matrix operand order repaired and checked by reading, not compiled, not run on hardware
//
#include <hip/hip_runtime.h>

#define DIM 512
#define NCLS 2048
#define BATCH 16384
#define BM 128
#define BN 128
#define BK 32
// LDS row stride: 40 bf16 (32 data + 8 pad) = 80 bytes = 5 v8bf chunks -> conflict-friendly, 16B aligned
#define LDV 5

typedef __attribute__((ext_vector_type(16))) __bf16          v16bf;
typedef __attribute__((ext_vector_type(8)))  __bf16          v8bf;
typedef __attribute__((ext_vector_type(8)))  unsigned short  v8us;
typedef __attribute__((ext_vector_type(8)))  float           v8f;

union FragU { v16bf v; v8bf q[2]; };
union V8U   { v8bf  b; v8us u; };

// Split fp32 into hi (truncated bf16, so v - hi is exact) + lo (RNE bf16 of residual).
__device__ __forceinline__ void split_bf16(float v, unsigned short& hs, unsigned short& ls) {
    unsigned u = __float_as_uint(v);
    hs = (unsigned short)(u >> 16);                       // round-toward-zero hi
    float hf = __uint_as_float((unsigned)hs << 16);
    float l  = v - hf;                                    // exact
    unsigned lu = __float_as_uint(l);
    ls = (unsigned short)((lu + 0x7FFFu + ((lu >> 16) & 1u)) >> 16);  // RNE lo
}

// ---------------- Kernel 1: per-class squared norms ----------------
__global__ __launch_bounds__(256) void kmeans_csq_kernel(const float* __restrict__ c,
                                                         float* __restrict__ csq) {
    const int lane = threadIdx.x & 31;
    const int cls  = blockIdx.x * 8 + (threadIdx.x >> 5);   // one wave per class
    const float4* p4 = (const float4*)(c + (size_t)cls * DIM);
    float s = 0.f;
#pragma unroll
    for (int j = 0; j < 4; ++j) {
        float4 q = p4[lane + j * 32];
        s += q.x * q.x + q.y * q.y + q.z * q.z + q.w * q.w;
    }
#pragma unroll
    for (int off = 16; off > 0; off >>= 1) s += __shfl_xor(s, off, 32);
    if (lane == 0) csq[cls] = s;
}

// ---------------- Kernel 2: logits = x . c^T - 0.5*||c||^2 (split-bf16 WMMA GEMM) --------
// Software-pipelined: global loads for chunk k+1 are issued before the WMMA burst of
// chunk k, so VMEM latency overlaps matrix compute.
__global__ __launch_bounds__(256) void kmeans_gemm_logits_kernel(
    const float* __restrict__ x, const float* __restrict__ c,
    const float* __restrict__ csq, float* __restrict__ out) {

    __shared__ v8bf sXhi[BM * LDV];
    __shared__ v8bf sXlo[BM * LDV];
    __shared__ v8bf sChi[BN * LDV];
    __shared__ v8bf sClo[BN * LDV];

    const int tid  = threadIdx.x;
    const int bn   = blockIdx.x;          // 0..15   (N tiles)
    const int bm   = blockIdx.y;          // 0..127  (M tiles)
    const int lane = tid & 31;
    const int w    = tid >> 5;            // 0..7
    const int wm   = w & 3;               // 4 wave-rows  -> 32 rows each
    const int wn   = w >> 2;              // 2 wave-cols  -> 64 cols each
    const int h    = lane >> 4;           // lane half
    const int lm   = lane & 15;

    const float* xg = x + (size_t)bm * BM * DIM;
    const float* cg = c + (size_t)bn * BN * DIM;

    v8f acc[2][4] = {};

    // register staging for the software pipeline
    float4 rx[2][2], rc[2][2];

    auto load_tile = [&](int k0) {
#pragma unroll
        for (int i = 0; i < 2; ++i) {
            int g   = tid + i * 256;      // 512 segments of 8 floats = 128x32
            int row = g >> 2;
            int seg = g & 3;
            const float* px = xg + (size_t)row * DIM + k0 + seg * 8;
            const float* pc = cg + (size_t)row * DIM + k0 + seg * 8;
            rx[i][0] = *(const float4*)px;
            rx[i][1] = *(const float4*)(px + 4);
            rc[i][0] = *(const float4*)pc;
            rc[i][1] = *(const float4*)(pc + 4);
        }
    };

    auto store_tile = [&]() {
#pragma unroll
        for (int i = 0; i < 2; ++i) {
            int g   = tid + i * 256;
            int row = g >> 2;
            int seg = g & 3;
            float xs[8] = {rx[i][0].x, rx[i][0].y, rx[i][0].z, rx[i][0].w,
                           rx[i][1].x, rx[i][1].y, rx[i][1].z, rx[i][1].w};
            float cs[8] = {rc[i][0].x, rc[i][0].y, rc[i][0].z, rc[i][0].w,
                           rc[i][1].x, rc[i][1].y, rc[i][1].z, rc[i][1].w};
            V8U xh, xl, ch, cl;
#pragma unroll
            for (int j = 0; j < 8; ++j) {
                unsigned short hs, ls;
                split_bf16(xs[j], hs, ls); xh.u[j] = hs; xl.u[j] = ls;
                split_bf16(cs[j], hs, ls); ch.u[j] = hs; cl.u[j] = ls;
            }
            sXhi[row * LDV + seg] = xh.b;  sXlo[row * LDV + seg] = xl.b;
            sChi[row * LDV + seg] = ch.b;  sClo[row * LDV + seg] = cl.b;
        }
    };

    load_tile(0);   // prologue

    for (int k0 = 0; k0 < DIM; k0 += BK) {
        // ---- commit staged chunk k0 into LDS ----
        store_tile();
        __syncthreads();

        // ---- issue global loads for the NEXT chunk (overlaps with WMMAs below) ----
        if (k0 + BK < DIM) load_tile(k0 + BK);

        // ---- load fragments (ISA VGPR layouts) ----
        v16bf ahi[2], alo[2], bhi[4], blo[4];
#pragma unroll
        for (int mi = 0; mi < 2; ++mi) {
            int r = wm * 32 + mi * 16 + lm;
            FragU fh, fl;
            fh.q[0] = sXhi[r * LDV + h];     fh.q[1] = sXhi[r * LDV + 2 + h];
            fl.q[0] = sXlo[r * LDV + h];     fl.q[1] = sXlo[r * LDV + 2 + h];
            ahi[mi] = fh.v;  alo[mi] = fl.v;
        }
#pragma unroll
        for (int ni = 0; ni < 4; ++ni) {
            int cn = wn * 64 + ni * 16 + lm;
            FragU fh, fl;
            fh.q[0] = sChi[cn * LDV + h];  fh.q[1] = sChi[cn * LDV + 2 + h];
            fl.q[0] = sClo[cn * LDV + h];  fl.q[1] = sClo[cn * LDV + 2 + h];
            bhi[ni] = fh.v;  blo[ni] = fl.v;
        }

        // ---- split-bf16 WMMA: hi*hi + hi*lo + lo*hi ----
#pragma unroll
        for (int mi = 0; mi < 2; ++mi)
#pragma unroll
            for (int ni = 0; ni < 4; ++ni) {
                acc[mi][ni] = __builtin_amdgcn_wmma_f32_16x16x32_bf16(
                    false, ahi[mi], false, bhi[ni], (short)0, acc[mi][ni], false, false);
                acc[mi][ni] = __builtin_amdgcn_wmma_f32_16x16x32_bf16(
                    false, ahi[mi], false, blo[ni], (short)0, acc[mi][ni], false, false);
                acc[mi][ni] = __builtin_amdgcn_wmma_f32_16x16x32_bf16(
                    false, alo[mi], false, bhi[ni], (short)0, acc[mi][ni], false, false);
            }
        __syncthreads();
    }

    // ---- epilogue: logits = acc - 0.5*||c||^2 ----
#pragma unroll
    for (int ni = 0; ni < 4; ++ni) {
        int col = bn * BN + wn * 64 + ni * 16 + lm;
        float bias = -0.5f * csq[col];
#pragma unroll
        for (int mi = 0; mi < 2; ++mi) {
            int rbase = bm * BM + wm * 32 + mi * 16 + h * 8;
            v8f a = acc[mi][ni];
#pragma unroll
            for (int j = 0; j < 8; ++j)
                out[(size_t)(rbase + j) * NCLS + col] = a[j] + bias;
        }
    }
}

// ---------------- Kernel 3: in-place row softmax over K=2048 ----------------
__global__ __launch_bounds__(256) void kmeans_softmax_kernel(float* __restrict__ out) {
    const int tid = threadIdx.x;
    float* p = out + (size_t)blockIdx.x * NCLS;
    const int w = tid >> 5, lane = tid & 31;
    __shared__ float smax[8];
    __shared__ float ssum[8];

    float v[8];
    float m = -3.4e38f;
#pragma unroll
    for (int i = 0; i < 8; ++i) { v[i] = p[tid + (i << 8)]; m = fmaxf(m, v[i]); }
#pragma unroll
    for (int off = 16; off > 0; off >>= 1) m = fmaxf(m, __shfl_xor(m, off, 32));
    if (lane == 0) smax[w] = m;
    __syncthreads();
    float t = smax[lane & 7];
#pragma unroll
    for (int off = 4; off > 0; off >>= 1) t = fmaxf(t, __shfl_xor(t, off, 32));
    m = t;  // every lane now holds the row max

    float s = 0.f;
#pragma unroll
    for (int i = 0; i < 8; ++i) { v[i] = __expf(v[i] - m); s += v[i]; }
#pragma unroll
    for (int off = 16; off > 0; off >>= 1) s += __shfl_xor(s, off, 32);
    if (lane == 0) ssum[w] = s;
    __syncthreads();
    float ts = ssum[lane & 7];
#pragma unroll
    for (int off = 4; off > 0; off >>= 1) ts += __shfl_xor(ts, off, 32);

    float inv = 1.0f / ts;
#pragma unroll
    for (int i = 0; i < 8; ++i) p[tid + (i << 8)] = v[i] * inv;
}

extern "C" void kernel_launch(void* const* d_in, const int* in_sizes, int n_in,
                              void* d_out, int out_size, void* d_ws, size_t ws_size,
                              hipStream_t stream) {
    (void)in_sizes; (void)n_in; (void)out_size; (void)ws_size;
    const float* x       = (const float*)d_in[0];
    const float* centers = (const float*)d_in[1];
    float* out = (float*)d_out;
    float* csq = (float*)d_ws;   // 2048 floats

    kmeans_csq_kernel<<<NCLS / 8, 256, 0, stream>>>(centers, csq);

    dim3 grid(NCLS / BN, BATCH / BM);   // (16, 128)
    kmeans_gemm_logits_kernel<<<grid, 256, 0, stream>>>(x, centers, csq, out);

    kmeans_softmax_kernel<<<BATCH, 256, 0, stream>>>(out);
}
